// RegionProposalNetwork_15831249453407
// MI455X (gfx1250) — matrix operand order repaired and checked by reading, not compiled, not run on hardware
//
#include <hip/hip_runtime.h>
#include <hip/hip_bf16.h>

// ---------------------------------------------------------------------------
// RPN forward for MI455X (gfx1250, wave32, WMMA).
// conv3x3 (512->512) as implicit GEMM M=40000,K=4608,N=512 via
// V_WMMA_F32_16X16X32_BF16 with double-buffered GLOBAL_LOAD_ASYNC_TO_LDS;
// 1x1 heads as GEMM M=40000,K=512,N=64 (45 used).
// Then anchor decode + per-image radix top-512 + bitonic sort + NMS + top-128.
// ---------------------------------------------------------------------------

#define B_     16
#define C_     512
#define H_     50
#define W_     50
#define HP     52
#define WP     52
#define F_     512
#define KK     4608          // 512*9
#define M_TOT  40000         // 16*50*50
#define NANCH  22500         // 50*50*9
#define PRE    512
#define POST   128

typedef float  v8f   __attribute__((ext_vector_type(8)));
typedef __bf16 v16bf __attribute__((ext_vector_type(16)));

struct __attribute__((aligned(16))) U4 { unsigned x, y, z, w; };
struct U4x2 { U4 lo, hi; };

static __device__ __forceinline__ v16bf frag_from(U4 lo, U4 hi) {
  U4x2 u; u.lo = lo; u.hi = hi;
  return __builtin_bit_cast(v16bf, u);   // 32B -> 16 x bf16
}

static __device__ __forceinline__ unsigned short f2bf(float f) {
  unsigned u = __float_as_uint(f);
  unsigned r = ((u >> 16) & 1u) + 0x7FFFu;      // round-to-nearest-even
  return (unsigned short)((u + r) >> 16);
}

// CDNA5 async copy: global memory -> LDS, tracked with ASYNCcnt.
// lds_off is the wave-relative LDS byte offset (low 32 bits of the generic
// shared-aperture address, per ISA 10.2).
static __device__ __forceinline__ void async_load_b128(unsigned lds_off,
                                                       const void* gaddr) {
  asm volatile("global_load_async_to_lds_b128 %0, %1, off"
               :: "v"(lds_off), "v"(gaddr)
               : "memory");
}

// ---------------------------------------------------------------------------
// 1) pack padded NHWC bf16 input: Ipad[b][yp][xp][c], 52x52 with zero halo
// ---------------------------------------------------------------------------
__global__ void pad_input_kernel(const float* __restrict__ fm,
                                 unsigned short* __restrict__ Ipad) {
  int idx = blockIdx.x * blockDim.x + threadIdx.x;
  const int total = B_ * HP * WP * C_;
  if (idx >= total) return;
  int c = idx & (C_ - 1);
  int t = idx >> 9;
  int xp = t % WP; t /= WP;
  int yp = t % HP;
  int b  = t / HP;
  float v = 0.0f;
  if (yp >= 1 && yp <= H_ && xp >= 1 && xp <= W_)
    v = fm[(((size_t)b * C_ + c) * H_ + (yp - 1)) * W_ + (xp - 1)];
  Ipad[idx] = f2bf(v);
}

// Wt[n][k], k = (ky*3+kx)*512 + c  (B^T: K contiguous per output channel)
__global__ void pack_w1_kernel(const float* __restrict__ w1,
                               unsigned short* __restrict__ Wt) {
  int idx = blockIdx.x * blockDim.x + threadIdx.x;
  if (idx >= F_ * KK) return;
  int n = idx / KK, k = idx % KK;
  int g = k >> 9, c = k & 511;
  int ky = g / 3, kx = g % 3;
  Wt[idx] = f2bf(w1[(((size_t)n * C_ + c) * 3 + ky) * 3 + kx]);
}

// Wh[n][c], n: 0..8 = w2 rows, 9..44 = w3 rows, 45..63 = zero pad
__global__ void pack_wh_kernel(const float* __restrict__ w2,
                               const float* __restrict__ w3,
                               unsigned short* __restrict__ Wh) {
  int idx = blockIdx.x * blockDim.x + threadIdx.x;
  if (idx >= 64 * C_) return;
  int n = idx >> 9, c = idx & 511;
  float v = 0.0f;
  if (n < 9)       v = w2[(size_t)n * C_ + c];
  else if (n < 45) v = w3[(size_t)(n - 9) * C_ + c];
  Wh[idx] = f2bf(v);
}

// ---------------------------------------------------------------------------
// 2) conv3x3 implicit GEMM: block tile 128(M) x 128(N), 8 waves (4x2),
//    wave tile 32x64 = 2x4 WMMA frags (8 wmma / K-step).
//    K loop: 9 taps x 16 steps of 32; double-buffered async LDS tiles.
// ---------------------------------------------------------------------------
__global__ __launch_bounds__(256) void conv3x3_wmma_kernel(
    const unsigned short* __restrict__ Ipad,
    const unsigned short* __restrict__ Wt,
    const float* __restrict__ b1,
    unsigned short* __restrict__ Sbf) {
  // double-buffered, row stride 80B (16B aligned, bank-spread)
  __shared__ unsigned short As[2 * 128 * 40];
  __shared__ unsigned short Bs[2 * 128 * 40];
  const unsigned asOff = (unsigned)(size_t)&As[0];
  const unsigned bsOff = (unsigned)(size_t)&Bs[0];

  const int tid  = threadIdx.x;
  const int lane = tid & 31;
  const int wave = tid >> 5;
  const int wm   = wave >> 1;          // 0..3 -> M offset wm*32
  const int wn   = wave & 1;           // 0..1 -> N offset wn*64
  const int lrow = lane & 15;
  const int hi   = (lane >> 4) & 1;    // lane half selects K sub-block

  const int mblk = blockIdx.x;         // 0..312 (313*128 = 40064, clamped)
  const int nblk = blockIdx.y;         // 0..3

  // A-tile loader: 128 rows x 4 x 16B -> 2 async loads/thread
  const int arow0 = tid >> 2;          // 0..63 (+64 second)
  const int aseg  = tid & 3;
  int am0 = mblk * 128 + arow0;        if (am0 >= M_TOT) am0 = M_TOT - 1;
  int am1 = mblk * 128 + arow0 + 64;   if (am1 >= M_TOT) am1 = M_TOT - 1;
  const unsigned short* aBase0 =
      Ipad + (((size_t)(am0 / 2500) * HP + (am0 % 2500) / 50) * WP +
              (am0 % 50)) * C_ + aseg * 8;
  const unsigned short* aBase1 =
      Ipad + (((size_t)(am1 / 2500) * HP + (am1 % 2500) / 50) * WP +
              (am1 % 50)) * C_ + aseg * 8;

  // B-tile loader: 128 cols x 4 x 16B -> 2 async loads/thread
  const int bcol0 = tid >> 2;          // 0..63 (+64 second)
  const int bseg  = tid & 3;
  const unsigned short* wB0 =
      Wt + (size_t)(nblk * 128 + bcol0) * KK + bseg * 8;
  const unsigned short* wB1 =
      Wt + (size_t)(nblk * 128 + bcol0 + 64) * KK + bseg * 8;

  const unsigned aLdsBase = asOff + (unsigned)(arow0 * 80 + aseg * 16);
  const unsigned bLdsBase = bsOff + (unsigned)(bcol0 * 80 + bseg * 16);

  auto issue_tiles = [&](int ks, int buf) {
    const int g   = ks >> 4;                          // 3x3 tap
    const int tap = ((g / 3) * WP + (g % 3)) * C_;
    const int kc  = (ks & 15) * 32;
    const unsigned bufOff = (unsigned)(buf * 10240);  // 128*40*2 bytes
    async_load_b128(aLdsBase + bufOff,               aBase0 + tap + kc);
    async_load_b128(aLdsBase + bufOff + 64 * 80,     aBase1 + tap + kc);
    async_load_b128(bLdsBase + bufOff,               wB0 + g * C_ + kc);
    async_load_b128(bLdsBase + bufOff + 64 * 80,     wB1 + g * C_ + kc);
  };

  v8f acc[2][4];
#pragma unroll
  for (int i = 0; i < 2; ++i)
#pragma unroll
    for (int j = 0; j < 4; ++j) acc[i][j] = (v8f)0.0f;

  issue_tiles(0, 0);

  for (int ks = 0; ks < 144; ++ks) {
    const int cur = ks & 1;
    __syncthreads();                         // back buffer free of readers
    if (ks + 1 < 144) {
      issue_tiles(ks + 1, cur ^ 1);
      asm volatile("s_wait_asynccnt 0x4" ::: "memory");  // batch ks landed
    } else {
      asm volatile("s_wait_asynccnt 0x0" ::: "memory");
    }
    __syncthreads();                         // all waves' tile data visible

    const unsigned short* Ab = As + cur * 5120;   // ushort units
    const unsigned short* Bb = Bs + cur * 5120;

    // A frag: lanes 0-15 K[0..7]+[16..23], lanes 16-31 K[8..15]+[24..31]
    v16bf aF[2], bF[4];
#pragma unroll
    for (int sm = 0; sm < 2; ++sm) {
      const unsigned short* p = Ab + (wm * 32 + sm * 16 + lrow) * 40 + hi * 8;
      aF[sm] = frag_from(*(const U4*)p, *(const U4*)(p + 16));
    }
    // B frag: lanes 0-15 K[0..7]+[16..23], lanes 16-31 K[8..15]+[24..31]
#pragma unroll
    for (int sn = 0; sn < 4; ++sn) {
      const unsigned short* p = Bb + (wn * 64 + sn * 16 + lrow) * 40 + hi * 8;
      bF[sn] = frag_from(*(const U4*)p, *(const U4*)(p + 16));
    }
#pragma unroll
    for (int sm = 0; sm < 2; ++sm)
#pragma unroll
      for (int sn = 0; sn < 4; ++sn)
        acc[sm][sn] = __builtin_amdgcn_wmma_f32_16x16x32_bf16(
            false, aF[sm], false, bF[sn], (short)0, acc[sm][sn], false, false);
  }

  // epilogue: +bias, ReLU, bf16 NHWC shared feature Sbf[m][n]
#pragma unroll
  for (int sn = 0; sn < 4; ++sn) {
    const int n = nblk * 128 + wn * 64 + sn * 16 + lrow;
    const float bias = b1[n];
#pragma unroll
    for (int sm = 0; sm < 2; ++sm) {
      const int mbase = mblk * 128 + wm * 32 + sm * 16 + hi * 8;
#pragma unroll
      for (int r = 0; r < 8; ++r) {
        const int m = mbase + r;
        if (m < M_TOT) {
          float v = fmaxf(acc[sm][sn][r] + bias, 0.0f);
          Sbf[(size_t)m * F_ + n] = f2bf(v);
        }
      }
    }
  }
}

// ---------------------------------------------------------------------------
// 3) head GEMM: M=40000 (tile 128), K=512, N=64. 8 waves (4x2), wave 32x32.
//    Staged loads + global_prefetch of the next K slice.
// ---------------------------------------------------------------------------
__global__ __launch_bounds__(256) void head_wmma_kernel(
    const unsigned short* __restrict__ Sbf,
    const unsigned short* __restrict__ Wh,
    float* __restrict__ Hout) {
  __shared__ unsigned short As[128 * 40];
  __shared__ unsigned short Bs[64 * 40];

  const int tid  = threadIdx.x;
  const int lane = tid & 31;
  const int wave = tid >> 5;
  const int wm   = wave >> 1;        // 0..3
  const int wn   = wave & 1;         // 0..1
  const int lrow = lane & 15;
  const int hi   = (lane >> 4) & 1;
  const int mblk = blockIdx.x;       // 0..312 (313*128 = 40064, clamped)

  const int arow = tid >> 2;         // 0..63 (+64 second)
  const int aseg = tid & 3;
  int am0 = mblk * 128 + arow;       if (am0 >= M_TOT) am0 = M_TOT - 1;
  int am1 = mblk * 128 + arow + 64;  if (am1 >= M_TOT) am1 = M_TOT - 1;
  const unsigned short* a0 = Sbf + (size_t)am0 * C_ + aseg * 8;
  const unsigned short* a1 = Sbf + (size_t)am1 * C_ + aseg * 8;

  const int bcol = tid >> 2;         // 0..63
  const unsigned short* wB = Wh + (size_t)bcol * C_ + aseg * 8;

  v8f acc[2][2];
#pragma unroll
  for (int i = 0; i < 2; ++i)
#pragma unroll
    for (int j = 0; j < 2; ++j) acc[i][j] = (v8f)0.0f;

  for (int ks = 0; ks < 16; ++ks) {
    const int kc = ks * 32;
    U4 av0 = *(const U4*)(a0 + kc);
    U4 av1 = *(const U4*)(a1 + kc);
    U4 bv  = *(const U4*)(wB + kc);
    if (ks + 1 < 16) {                       // global_prefetch_b8
      __builtin_prefetch(a0 + kc + 32, 0, 1);
      __builtin_prefetch(a1 + kc + 32, 0, 1);
    }

    __syncthreads();
    *(U4*)(As + arow * 40 + aseg * 8)        = av0;
    *(U4*)(As + (arow + 64) * 40 + aseg * 8) = av1;
    *(U4*)(Bs + bcol * 40 + aseg * 8)        = bv;
    __syncthreads();

    v16bf aF[2], bF[2];
#pragma unroll
    for (int sm = 0; sm < 2; ++sm) {
      const unsigned short* p = As + (wm * 32 + sm * 16 + lrow) * 40 + hi * 8;
      aF[sm] = frag_from(*(const U4*)p, *(const U4*)(p + 16));
    }
#pragma unroll
    for (int sn = 0; sn < 2; ++sn) {
      const unsigned short* p = Bs + (wn * 32 + sn * 16 + lrow) * 40 + hi * 8;
      bF[sn] = frag_from(*(const U4*)p, *(const U4*)(p + 16));
    }
#pragma unroll
    for (int sm = 0; sm < 2; ++sm)
#pragma unroll
      for (int sn = 0; sn < 2; ++sn)
        acc[sm][sn] = __builtin_amdgcn_wmma_f32_16x16x32_bf16(
            false, aF[sm], false, bF[sn], (short)0, acc[sm][sn], false, false);
  }

#pragma unroll
  for (int sn = 0; sn < 2; ++sn) {
    const int n = wn * 32 + sn * 16 + lrow;     // 0..63
#pragma unroll
    for (int sm = 0; sm < 2; ++sm) {
      const int mbase = mblk * 128 + wm * 32 + sm * 16 + hi * 8;
#pragma unroll
      for (int r = 0; r < 8; ++r) {
        const int m = mbase + r;
        if (m < M_TOT) Hout[(size_t)m * 64 + n] = acc[sm][sn][r];
      }
    }
  }
}

// ---------------------------------------------------------------------------
// 4) sigmoid + anchor decode  (anchor order: location-major, anchor-minor)
// ---------------------------------------------------------------------------
__global__ void decode_kernel(const float* __restrict__ Hout,
                              const float* __restrict__ b2,
                              const float* __restrict__ b3,
                              float* __restrict__ boxes,
                              float* __restrict__ scores) {
  int idx = blockIdx.x * blockDim.x + threadIdx.x;
  if (idx >= B_ * NANCH) return;
  int b   = idx / NANCH;
  int i   = idx % NANCH;
  int loc = i / 9;
  int a   = i % 9;
  int y = loc / 50, x = loc % 50;
  const float* h = Hout + (size_t)(b * 2500 + loc) * 64;

  float s  = 1.0f / (1.0f + expf(-(h[a] + b2[a])));
  float dx = h[9 + a * 4 + 0] + b3[a * 4 + 0];
  float dy = h[9 + a * 4 + 1] + b3[a * 4 + 1];
  float dw = h[9 + a * 4 + 2] + b3[a * 4 + 2];
  float dh = h[9 + a * 4 + 3] + b3[a * 4 + 3];

  const float SZ[3] = {32.0f, 64.0f, 128.0f};
  const float RT[3] = {0.5f, 1.0f, 2.0f};
  float sz = SZ[a / 3];
  float sq = sqrtf(RT[a % 3]);
  float ah = sz * sq;
  float aw = sz / sq;
  float cx = (x + 0.5f) * 16.0f;
  float cy = (y + 0.5f) * 16.0f;

  float px = cx + dx * aw;
  float py = cy + dy * ah;
  float pw = aw * expf(dw);
  float ph = ah * expf(dh);

  float* o = boxes + (size_t)idx * 4;
  o[0] = px - 0.5f * pw;
  o[1] = py - 0.5f * ph;
  o[2] = px + 0.5f * pw;
  o[3] = py + 0.5f * ph;
  scores[idx] = s;
}

// ---------------------------------------------------------------------------
// 5) per-image: radix top-512 -> bitonic sort -> clip/valid -> NMS -> top-128
//    scores are sigmoid outputs > 0 => float bits are order-preserving.
// ---------------------------------------------------------------------------
__global__ __launch_bounds__(1024) void proposals_kernel(
    const float* __restrict__ boxes,
    const float* __restrict__ scores,
    float* __restrict__ out_b,
    float* __restrict__ out_s) {
  const int img = blockIdx.x;
  const int tid = threadIdx.x;
  const int NT  = 1024;
  const float* sc = scores + (size_t)img * NANCH;
  const float* bx = boxes  + (size_t)img * NANCH * 4;

  __shared__ unsigned hist[256];
  __shared__ unsigned s_prefix;
  __shared__ int s_need, cntG, cntE, s_total;
  __shared__ float candKey[PRE];
  __shared__ int   candIdx[PRE];
  __shared__ float cbx[PRE][4];
  __shared__ float carea[PRE];
  __shared__ int   keepf[PRE];
  __shared__ int   rankArr[PRE];

  // ---- radix select: find 512th-largest key (MSB-first 8-bit digits) ----
  unsigned prefix = 0;
  int need = PRE;
  for (int shift = 24; shift >= 0; shift -= 8) {
    for (int j = tid; j < 256; j += NT) hist[j] = 0;
    __syncthreads();
    unsigned maskHi = (shift == 24) ? 0u : (0xFFFFFFFFu << (shift + 8));
    for (int i = tid; i < NANCH; i += NT) {
      unsigned k = __float_as_uint(sc[i]);
      if ((k & maskHi) == (prefix & maskHi))
        atomicAdd(&hist[(k >> shift) & 255], 1u);
    }
    __syncthreads();
    if (tid == 0) {
      int cum = 0, bsel = 0;
      for (int bin = 255; bin >= 0; --bin) {
        int c = (int)hist[bin];
        if (cum + c >= need) { bsel = bin; break; }
        cum += c;
      }
      s_prefix = prefix | ((unsigned)bsel << shift);
      s_need = need - cum;
    }
    __syncthreads();
    prefix = s_prefix;
    need   = s_need;
    __syncthreads();
  }
  const unsigned T = prefix;          // exact 32-bit threshold key
  const int nGreater = PRE - need;

  if (tid == 0) { cntG = 0; cntE = 0; }
  __syncthreads();
  for (int i = tid; i < NANCH; i += NT) {
    unsigned k = __float_as_uint(sc[i]);
    if (k > T) {
      int p = atomicAdd(&cntG, 1);
      candKey[p] = sc[i]; candIdx[p] = i;
    } else if (k == T) {
      int q = atomicAdd(&cntE, 1);
      if (q < need) {
        int p = nGreater + q;
        candKey[p] = sc[i]; candIdx[p] = i;
      }
    }
  }
  __syncthreads();

  // ---- bitonic sort 512, descending by key, ascending idx on ties ----
  for (int kk = 2; kk <= PRE; kk <<= 1) {
    for (int j = kk >> 1; j > 0; j >>= 1) {
      for (int i = tid; i < PRE; i += NT) {
        int ixj = i ^ j;
        if (ixj > i) {
          bool dirDesc = ((i & kk) == 0);
          float a = candKey[i], b = candKey[ixj];
          int ia = candIdx[i], ib = candIdx[ixj];
          bool aFirst = (a > b) || (a == b && ia < ib);
          bool doSwap = dirDesc ? (!aFirst) : aFirst;
          if (doSwap) {
            candKey[i] = b; candKey[ixj] = a;
            candIdx[i] = ib; candIdx[ixj] = ia;
          }
        }
      }
      __syncthreads();
    }
  }

  // ---- clip + validity ----
  for (int i = tid; i < PRE; i += NT) {
    int idx = candIdx[i];
    float x1 = fminf(fmaxf(bx[idx * 4 + 0], 0.0f), 800.0f);
    float y1 = fminf(fmaxf(bx[idx * 4 + 1], 0.0f), 800.0f);
    float x2 = fminf(fmaxf(bx[idx * 4 + 2], 0.0f), 800.0f);
    float y2 = fminf(fmaxf(bx[idx * 4 + 3], 0.0f), 800.0f);
    cbx[i][0] = x1; cbx[i][1] = y1; cbx[i][2] = x2; cbx[i][3] = y2;
    float w = x2 - x1, h = y2 - y1;
    carea[i] = w * h;
    keepf[i] = (w >= 0.001f && h >= 0.001f && candKey[i] >= 0.5f) ? 1 : 0;
  }
  __syncthreads();

  // ---- greedy NMS over sorted candidates ----
  for (int i = 0; i < PRE; ++i) {
    if (keepf[i]) {
      float x1 = cbx[i][0], y1 = cbx[i][1], x2 = cbx[i][2], y2 = cbx[i][3];
      float ar = carea[i];
      for (int j = i + 1 + tid; j < PRE; j += NT) {
        if (keepf[j]) {
          float lx = fmaxf(x1, cbx[j][0]), ly = fmaxf(y1, cbx[j][1]);
          float rx = fminf(x2, cbx[j][2]), ry = fminf(y2, cbx[j][3]);
          float iw = fmaxf(rx - lx, 0.0f), ih = fmaxf(ry - ly, 0.0f);
          float inter = iw * ih;
          float iou = inter / (ar + carea[j] - inter + 1e-9f);
          if (iou > 0.7f) keepf[j] = 0;
        }
      }
    }
    __syncthreads();
  }

  // ---- compact kept (already score-desc) into first 128 slots ----
  if (tid == 0) {
    int c = 0;
    for (int i = 0; i < PRE; ++i) { rankArr[i] = c; c += keepf[i]; }
    s_total = c;
  }
  __syncthreads();
  const int total = s_total;
  for (int r = tid; r < POST; r += NT) {
    if (r >= total) {
      out_s[img * POST + r] = 0.0f;
      out_b[(img * POST + r) * 4 + 0] = 0.0f;
      out_b[(img * POST + r) * 4 + 1] = 0.0f;
      out_b[(img * POST + r) * 4 + 2] = 0.0f;
      out_b[(img * POST + r) * 4 + 3] = 0.0f;
    }
  }
  for (int i = tid; i < PRE; i += NT) {
    if (keepf[i]) {
      int r = rankArr[i];
      if (r < POST) {
        out_s[img * POST + r] = candKey[i];
        out_b[(img * POST + r) * 4 + 0] = cbx[i][0];
        out_b[(img * POST + r) * 4 + 1] = cbx[i][1];
        out_b[(img * POST + r) * 4 + 2] = cbx[i][2];
        out_b[(img * POST + r) * 4 + 3] = cbx[i][3];
      }
    }
  }
}

// ---------------------------------------------------------------------------
extern "C" void kernel_launch(void* const* d_in, const int* in_sizes, int n_in,
                              void* d_out, int out_size, void* d_ws, size_t ws_size,
                              hipStream_t stream) {
  const float* fm = (const float*)d_in[0];
  const float* w1 = (const float*)d_in[1];
  const float* b1 = (const float*)d_in[2];
  const float* w2 = (const float*)d_in[3];
  const float* b2 = (const float*)d_in[4];
  const float* w3 = (const float*)d_in[5];
  const float* b3 = (const float*)d_in[6];

  char* ws = (char*)d_ws;
  size_t off = 0;
  auto alloc = [&](size_t bytes) {
    size_t o = off;
    off += (bytes + 255) & ~(size_t)255;
    return o;
  };
  unsigned short* Ipad = (unsigned short*)(ws + alloc((size_t)B_ * HP * WP * C_ * 2));   // 44.3 MB
  unsigned short* Wt   = (unsigned short*)(ws + alloc((size_t)F_ * KK * 2));             //  4.7 MB
  unsigned short* Wh   = (unsigned short*)(ws + alloc((size_t)64 * C_ * 2));             //  64 KB
  unsigned short* Sbf  = (unsigned short*)(ws + alloc((size_t)M_TOT * F_ * 2));          // 41.0 MB
  float*          Hout = (float*)(ws + alloc((size_t)M_TOT * 64 * 4));                   // 10.2 MB
  float*          Boxes= (float*)(ws + alloc((size_t)B_ * NANCH * 4 * 4));               //  5.8 MB
  float*          Scor = (float*)(ws + alloc((size_t)B_ * NANCH * 4));                   //  1.4 MB

  float* outB = (float*)d_out;                 // [16,128,4]
  float* outS = outB + (size_t)B_ * POST * 4;  // [16,128]

  {
    int n = B_ * HP * WP * C_;
    pad_input_kernel<<<(n + 255) / 256, 256, 0, stream>>>(fm, Ipad);
  }
  {
    int n = F_ * KK;
    pack_w1_kernel<<<(n + 255) / 256, 256, 0, stream>>>(w1, Wt);
  }
  {
    int n = 64 * C_;
    pack_wh_kernel<<<(n + 255) / 256, 256, 0, stream>>>(w2, w3, Wh);
  }
  conv3x3_wmma_kernel<<<dim3(313, 4), 256, 0, stream>>>(Ipad, Wt, b1, Sbf);
  head_wmma_kernel<<<dim3(313), 256, 0, stream>>>(Sbf, Wh, Hout);
  {
    int n = B_ * NANCH;
    decode_kernel<<<(n + 255) / 256, 256, 0, stream>>>(Hout, b2, b3, Boxes, Scor);
  }
  proposals_kernel<<<B_, 1024, 0, stream>>>(Boxes, Scor, outB, outS);
}
